// DebugStepWiseLinearTransformerLayer_57724360458871
// MI455X (gfx1250) — matrix operand order repaired and checked by reading, not compiled, not run on hardware
//
#include <hip/hip_runtime.h>
#include <cstdint>

typedef __attribute__((ext_vector_type(16))) __bf16 v16bf;
typedef __attribute__((ext_vector_type(8)))  float  v8f;

#define D_MODEL 512
#define N_HEAD  8
#define D_HEAD  64
#define SEQ     2048
#define BATCH   16
#define ROWS    (SEQ * BATCH)      // 32768
#define QKV_N   (3 * D_MODEL)      // 1536
#define T_CHUNK 32
#define SCALE_Q 0.125f             // 1/sqrt(64)

static __device__ __forceinline__ v8f wmma_bf16(v16bf a, v16bf b, v8f c) {
  // D = A(16x32 bf16) * B(32x16 bf16) + C(16x16 f32)
  return __builtin_amdgcn_wmma_f32_16x16x32_bf16(false, a, false, b, (short)0, c,
                                                 false, false);
}

// Build a 16-element bf16 fragment from two 16B-aligned 8-element runs.
// Forces b128 loads (ds_load_b128 / global_load_b128) instead of scalar gathers.
static __device__ __forceinline__ v16bf ld_frag(const __bf16* p0, const __bf16* p1) {
  union { v16bf v; uint4 q[2]; } u;
  u.q[0] = *(const uint4*)p0;
  u.q[1] = *(const uint4*)p1;
  return u.v;
}

// ---------------------------------------------------------------- prep kernels
__global__ void cvt_f32_to_bf16(const float* __restrict__ src,
                                __bf16* __restrict__ dst, long n) {
  long i = (long)blockIdx.x * blockDim.x + threadIdx.x;
  if (i < n) dst[i] = (__bf16)src[i];
}

// Swizzle weights into per-lane B-fragment order:
//   tile block (nt = n>>4, kt = k>>5) stored at ((nt*(K/32)+kt)*512);
//   within block: lane = (n&15) + 16*((k>>4)&1), elem = k&15  -> lane*16+elem.
// A B-fragment load is then one contiguous 32B per lane (2x uint4).
__global__ void prep_wqkv(const float* __restrict__ Wq,
                          const float* __restrict__ Wkv,
                          __bf16* __restrict__ Wt) {
  long i = (long)blockIdx.x * blockDim.x + threadIdx.x;
  if (i >= (long)D_MODEL * QKV_N) return;
  int k = (int)(i / QKV_N), n = (int)(i % QKV_N);
  float v = (n < D_MODEL) ? Wq[(size_t)n * D_MODEL + k]
                          : Wkv[(size_t)(n - D_MODEL) * D_MODEL + k];
  int nt = n >> 4, kt = k >> 5;
  int lane = (n & 15) + (((k >> 4) & 1) << 4);
  size_t dst = (((size_t)nt * (D_MODEL >> 5) + kt) << 9) + ((size_t)lane << 4) + (k & 15);
  Wt[dst] = (__bf16)v;
}

__global__ void prep_wo(const float* __restrict__ Wo, __bf16* __restrict__ Wt) {
  long i = (long)blockIdx.x * blockDim.x + threadIdx.x;
  if (i >= (long)D_MODEL * D_MODEL) return;
  int k = (int)(i / D_MODEL), n = (int)(i % D_MODEL);
  int nt = n >> 4, kt = k >> 5;
  int lane = (n & 15) + (((k >> 4) & 1) << 4);
  size_t dst = (((size_t)nt * (D_MODEL >> 5) + kt) << 9) + ((size_t)lane << 4) + (k & 15);
  Wt[dst] = (__bf16)Wo[(size_t)n * D_MODEL + k];
}

// ---------------------------------------------------------------- WMMA GEMM
// Y[M,N] = X[M,K] @ W(swizzled).  No LDS: A-fragments are direct b128 global
// loads from row-major X; B-fragments are direct b128 loads from the swizzled
// weight buffer (weights stay resident in L2).  Block = 128 thr = 4 waves
// stacked in M; block tile 128x64, wave tile 32x64 (2x4 WMMA frags).
__global__ void gemm_bf16(const __bf16* __restrict__ X,
                          const __bf16* __restrict__ Wsw,
                          __bf16* __restrict__ Ybf, float* __restrict__ Yf,
                          int N, int K, int elu_ncut) {
  const int tid = threadIdx.x, wave = tid >> 5, lane = tid & 31;
  const int l15 = lane & 15;
  const int k0a = (lane < 16) ? 0 : 16;   // A-fragment K base
  const int mb  = (lane < 16) ? 0 : 8;    // C/D row base
  const int m0 = blockIdx.y * 128 + wave * 32;
  const int n0 = blockIdx.x * 64;
  const int kt_cnt = K >> 5;

  v8f acc[2][4] = {};
  const __bf16* xrow0 = X + (size_t)(m0 + l15) * K + k0a;
  const __bf16* xrow1 = xrow0 + (size_t)16 * K;
  const __bf16* wp = Wsw + ((size_t)(n0 >> 4) * kt_cnt << 9) + ((size_t)lane << 4);

  for (int kt = 0; kt < kt_cnt; ++kt) {
    if (kt + 2 < kt_cnt) {  // L2/L0 prefetch of next-next K slab
      __builtin_prefetch(xrow0 + (size_t)(kt + 2) * 32, 0, 1);
      __builtin_prefetch(xrow1 + (size_t)(kt + 2) * 32, 0, 1);
    }
    v16bf a0 = ld_frag(xrow0 + (size_t)kt * 32, xrow0 + (size_t)kt * 32 + 8);
    v16bf a1 = ld_frag(xrow1 + (size_t)kt * 32, xrow1 + (size_t)kt * 32 + 8);
    v16bf b[4];
#pragma unroll
    for (int tj = 0; tj < 4; ++tj) {
      const __bf16* bp = wp + (((size_t)tj * kt_cnt + kt) << 9);
      b[tj] = ld_frag(bp, bp + 8);
    }
#pragma unroll
    for (int tj = 0; tj < 4; ++tj) {
      acc[0][tj] = wmma_bf16(a0, b[tj], acc[0][tj]);
      acc[1][tj] = wmma_bf16(a1, b[tj], acc[1][tj]);
    }
  }

  // epilogue (branchless elu(x)+1 on n < elu_ncut when storing bf16)
#pragma unroll
  for (int ti = 0; ti < 2; ++ti) {
#pragma unroll
    for (int tj = 0; tj < 4; ++tj) {
      int n = n0 + tj * 16 + l15;
#pragma unroll
      for (int r = 0; r < 8; ++r) {
        int m = m0 + ti * 16 + mb + r;
        float v = acc[ti][tj][r];
        if (Ybf) {
          float ev = __expf(v);                      // elu+1, x<=0 branch
          float e1 = (v > 0.0f) ? (v + 1.0f) : ev;   // v_cndmask
          v = (n < elu_ncut) ? e1 : v;
          Ybf[(size_t)m * N + n] = (__bf16)v;
        } else {
          Yf[(size_t)m * N + n] = v;
        }
      }
    }
  }
}

// ---------------------------------------------------------------- chunked scan
// One block per (b, head).  Fast-weight state S (64x64 f32) lives in LDS.
// Per T=32 chunk:
//   Attn = tril(Q K^T)          Out = Attn V + Q S_prev^T       S += V^T K
// All WMMA fragments read contiguous LDS runs (b128): K is staged both
// row-major (Kc) and transposed (KT); V only transposed (VT); state has a
// straight bf16 shadow (Sb) so B = S^T reads Sb[n][k] contiguously.
__global__ void scan_kernel(const __bf16* __restrict__ qkv,
                            __bf16* __restrict__ layer) {
  __shared__ float Sst[64][65];                        // f32 state S[i][j]
  __shared__ __align__(16) __bf16 Sb[64][80];          // bf16 copy of S
  __shared__ __align__(16) __bf16 Qc[T_CHUNK][80];     // Q row-major
  __shared__ __align__(16) __bf16 Kc[T_CHUNK][80];     // K row-major
  __shared__ __align__(16) __bf16 KT[64][40];          // K^T
  __shared__ __align__(16) __bf16 VT[64][40];          // V^T
  __shared__ __align__(16) __bf16 At[T_CHUNK][40];     // masked attn

  const int bh = blockIdx.x;           // 0..127
  const int b = bh >> 3, hh = bh & 7;
  const int tid = threadIdx.x, wave = tid >> 5, lane = tid & 31;
  const int l15 = lane & 15;
  const int k0a = (lane < 16) ? 0 : 16;
  const int kbb = (lane < 16) ? 0 : 16;
  const int mb  = (lane < 16) ? 0 : 8;
  const size_t qcol = (size_t)hh * D_HEAD;

  for (int i = tid; i < 64 * 64; i += 128) Sst[i >> 6][i & 63] = 0.0f;
  __syncthreads();

  for (int c = 0; c < SEQ / T_CHUNK; ++c) {
    {   // stage Q,K,V: 32 rows x 64 cols each; 4 threads/row, 16 bf16 each
      int t = tid >> 2, cc = (tid & 3) * 16;
      size_t m = (size_t)(c * T_CHUNK + t) * BATCH + b;
      const __bf16* rowp = qkv + m * QKV_N;

      // Q and K row-major slabs: async global->LDS DMA (ASYNCcnt path)
      uint32_t ldsq = (uint32_t)(uintptr_t)&Qc[t][cc];
      uint64_t gq   = (uint64_t)(uintptr_t)(rowp + qcol + cc);
      asm volatile("global_load_async_to_lds_b128 %0, %1, off"
                   :: "v"(ldsq), "v"(gq) : "memory");
      asm volatile("global_load_async_to_lds_b128 %0, %1, off"
                   :: "v"((uint32_t)(ldsq + 16)), "v"((uint64_t)(gq + 16)) : "memory");
      uint32_t ldsk = (uint32_t)(uintptr_t)&Kc[t][cc];
      uint64_t gk   = (uint64_t)(uintptr_t)(rowp + D_MODEL + qcol + cc);
      asm volatile("global_load_async_to_lds_b128 %0, %1, off"
                   :: "v"(ldsk), "v"(gk) : "memory");
      asm volatile("global_load_async_to_lds_b128 %0, %1, off"
                   :: "v"((uint32_t)(ldsk + 16)), "v"((uint64_t)(gk + 16)) : "memory");

      // K^T via registers (reads same cachelines the async DMA touches)
      union { uint4 q[2]; __bf16 e[16]; } kk;
      kk.q[0] = *(const uint4*)(rowp + D_MODEL + qcol + cc);
      kk.q[1] = *(const uint4*)(rowp + D_MODEL + qcol + cc + 8);
#pragma unroll
      for (int i2 = 0; i2 < 16; ++i2) KT[cc + i2][t] = kk.e[i2];

      // V^T (V is only ever consumed transposed)
      union { uint4 q[2]; __bf16 e[16]; } vv;
      vv.q[0] = *(const uint4*)(rowp + 2 * D_MODEL + qcol + cc);
      vv.q[1] = *(const uint4*)(rowp + 2 * D_MODEL + qcol + cc + 8);
#pragma unroll
      for (int i2 = 0; i2 < 16; ++i2) VT[cc + i2][t] = vv.e[i2];
    }
    // bf16 shadow of the state (straight copy; Q@S^T reads it as Sb[n][k])
    for (int i = tid; i < 64 * 64; i += 128)
      Sb[i >> 6][i & 63] = (__bf16)Sst[i >> 6][i & 63];
    asm volatile("s_wait_asynccnt 0x0" ::: "memory");
    __syncthreads();

    {   // Attn = tril(Q K^T): one 16x16 tile per wave
      int ti = wave >> 1, tj = wave & 1;
      v8f acc = {};
      int row = ti * 16 + l15, col = tj * 16 + l15;
#pragma unroll
      for (int ks = 0; ks < 64; ks += 32) {
        v16bf a = ld_frag(&Qc[row][ks + k0a], &Qc[row][ks + k0a + 8]);
        v16bf bb = ld_frag(&Kc[col][ks + kbb], &Kc[col][ks + kbb + 8]); // B=K^T
        acc = wmma_bf16(a, bb, acc);
      }
      int tbase = ti * 16 + mb, s = tj * 16 + l15;
#pragma unroll
      for (int r = 0; r < 8; ++r) {
        float v = (s <= tbase + r) ? acc[r] : 0.0f;   // causal (incl. diag)
        At[tbase + r][s] = (__bf16)v;
      }
    }
    __syncthreads();

    // Out tiles (32 x 64): 8 tiles, 2 per wave
#pragma unroll
    for (int j = 0; j < 2; ++j) {
      int idx = wave * 2 + j, ti = idx >> 2, nj = idx & 3;
      int row = ti * 16 + l15, col = nj * 16 + l15;
      v8f acc = {};
      {   // Attn @ V  (K = 32); B[k][n] = V[k][n] = VT[n][k]
        v16bf a = ld_frag(&At[row][k0a], &At[row][k0a + 8]);
        v16bf bb = ld_frag(&VT[col][kbb], &VT[col][kbb + 8]);
        acc = wmma_bf16(a, bb, acc);
      }
#pragma unroll
      for (int ks = 0; ks < 64; ks += 32) {   // Q @ S^T; B[k][n] = Sb[n][k]
        v16bf a = ld_frag(&Qc[row][ks + k0a], &Qc[row][ks + k0a + 8]);
        v16bf bb = ld_frag(&Sb[col][ks + kbb], &Sb[col][ks + kbb + 8]);
        acc = wmma_bf16(a, bb, acc);
      }
      int colg = hh * D_HEAD + nj * 16 + l15;
#pragma unroll
      for (int r = 0; r < 8; ++r) {
        int t = ti * 16 + mb + r;
        size_t m = (size_t)(c * T_CHUNK + t) * BATCH + b;
        layer[m * D_MODEL + colg] = (__bf16)(SCALE_Q * acc[r]);
      }
    }

    // State update S += V^T K: 16 tiles, 4 per wave
    //   A[i][t] = V[t][i] = VT[i][t]   B[t][j] = K[t][j] = KT[j][t]
#pragma unroll
    for (int q4 = 0; q4 < 4; ++q4) {
      int idx = wave * 4 + q4, ii = idx >> 2, jj = idx & 3;
      int irow = ii * 16 + l15, col = jj * 16 + l15;
      v8f acc;
#pragma unroll
      for (int r = 0; r < 8; ++r) acc[r] = Sst[ii * 16 + mb + r][col];
      v16bf a = ld_frag(&VT[irow][k0a], &VT[irow][k0a + 8]);
      v16bf bb = ld_frag(&KT[col][kbb], &KT[col][kbb + 8]);
      acc = wmma_bf16(a, bb, acc);
#pragma unroll
      for (int r = 0; r < 8; ++r) Sst[ii * 16 + mb + r][col] = acc[r];
    }
    __syncthreads();
  }
}

// ---------------------------------------------------------------- residual + LN
__global__ void ln_kernel(const float* __restrict__ h,
                          const float* __restrict__ gamma,
                          const float* __restrict__ beta,
                          float* __restrict__ out) {
  __shared__ float rs[128], rq[128];
  const int row = blockIdx.x, tid = threadIdx.x;
  const float* hr = h + (size_t)row * D_MODEL;
  float* orow = out + (size_t)row * D_MODEL;
  float x[4], s = 0.0f, s2 = 0.0f;
#pragma unroll
  for (int i = 0; i < 4; ++i) {
    int idx = i * 128 + tid;
    x[i] = hr[idx] + orow[idx];
    s += x[i]; s2 += x[i] * x[i];
  }
  rs[tid] = s; rq[tid] = s2;
  __syncthreads();
  for (int o = 64; o > 0; o >>= 1) {
    if (tid < o) { rs[tid] += rs[tid + o]; rq[tid] += rq[tid + o]; }
    __syncthreads();
  }
  float mu = rs[0] * (1.0f / D_MODEL);
  float var = rq[0] * (1.0f / D_MODEL) - mu * mu;
  float inv = rsqrtf(var + 1e-5f);
#pragma unroll
  for (int i = 0; i < 4; ++i) {
    int idx = i * 128 + tid;
    orow[idx] = (x[i] - mu) * inv * gamma[idx] + beta[idx];
  }
}

// ---------------------------------------------------------------- launch
extern "C" void kernel_launch(void* const* d_in, const int* in_sizes, int n_in,
                              void* d_out, int out_size, void* d_ws, size_t ws_size,
                              hipStream_t stream) {
  (void)in_sizes; (void)n_in; (void)out_size; (void)ws_size;
  const float* h     = (const float*)d_in[0];
  const float* Wq    = (const float*)d_in[1];
  const float* Wkv   = (const float*)d_in[2];
  const float* Wo    = (const float*)d_in[3];
  const float* gamma = (const float*)d_in[4];
  const float* beta  = (const float*)d_in[5];
  float* out = (float*)d_out;

  char* ws = (char*)d_ws;
  size_t off = 0;
  __bf16* hb    = (__bf16*)(ws + off); off += (size_t)ROWS * D_MODEL * 2;   // 32 MB
  __bf16* qkv   = (__bf16*)(ws + off); off += (size_t)ROWS * QKV_N * 2;     // 96 MB
  __bf16* layer = (__bf16*)(ws + off); off += (size_t)ROWS * D_MODEL * 2;   // 32 MB
  __bf16* wqkvt = (__bf16*)(ws + off); off += (size_t)D_MODEL * QKV_N * 2;
  __bf16* wot   = (__bf16*)(ws + off); off += (size_t)D_MODEL * D_MODEL * 2;

  long nh = (long)ROWS * D_MODEL;
  cvt_f32_to_bf16<<<(nh + 255) / 256, 256, 0, stream>>>(h, hb, nh);
  prep_wqkv<<<(D_MODEL * QKV_N + 255) / 256, 256, 0, stream>>>(Wq, Wkv, wqkvt);
  prep_wo<<<(D_MODEL * D_MODEL + 255) / 256, 256, 0, stream>>>(Wo, wot);

  // QKV projection with elu+1 epilogue on q,k columns (n < 1024)
  gemm_bf16<<<dim3(QKV_N / 64, ROWS / 128), 128, 0, stream>>>(
      hb, wqkvt, qkv, nullptr, QKV_N, D_MODEL, 2 * D_MODEL);

  // chunked fast-weight scan, one block per (batch, head)
  scan_kernel<<<BATCH * N_HEAD, 128, 0, stream>>>(qkv, layer);

  // output projection -> f32 attn directly into d_out
  gemm_bf16<<<dim3(D_MODEL / 64, ROWS / 128), 128, 0, stream>>>(
      layer, wot, nullptr, out, D_MODEL, D_MODEL, 0);

  // residual + LayerNorm in place
  ln_kernel<<<ROWS, 128, 0, stream>>>(h, gamma, beta, out);
}
